// ST_ATT_Layer_53678501265627
// MI455X (gfx1250) — matrix operand order repaired and checked by reading, not compiled, not run on hardware
//
#include <hip/hip_runtime.h>
#include <hip/hip_bf16.h>
#include <math.h>

typedef __attribute__((ext_vector_type(16))) _Float16 v16h;
typedef __attribute__((ext_vector_type(8)))  _Float16 v8h;
typedef __attribute__((ext_vector_type(8)))  float    v8f;
typedef __attribute__((ext_vector_type(4)))  unsigned int v4u;
typedef __attribute__((ext_vector_type(4)))  int v4i;
typedef __attribute__((ext_vector_type(8)))  int v8i;

#define F_IN  128
#define H_NUM 8
#define DH    64
#define T_LEN 8
#define J_NUM 22
#define S_LEN 66
#define W_NUM 6
#define B_SZ  128
#define SP    80    // S padded for M dims (5 x 16)
#define SK    96    // S padded for K dims (3 x 32)
#define NEGV  (-9e15f)

// workspace layout (in _Float16 elements)
#define YB_OFF   0                         // y intermediate: 16896 x 512
#define YB_SZ    (16896 * 512)
#define WQKV_OFF (YB_OFF + YB_SZ)          // WqkvT: [3][512 col][128 k]
#define WQKV_SZ  (3 * 512 * 128)
#define WCLS_OFF (WQKV_OFF + WQKV_SZ)      // WclsT: [32 j][96 k] zero-padded
#define WCLS_SZ  (32 * 96)
#define WFT_OFF  (WCLS_OFF + WCLS_SZ)      // WftT: [512 col][512 k]
#define WFT_SZ   (512 * 512)

__device__ __forceinline__ v8f wmma16(v16h a, v16h b, v8f c) {
  return __builtin_amdgcn_wmma_f32_16x16x32_f16(false, a, false, b, (short)0, c,
                                                false, false);
}

// A fragment: two aligned 16B chunks at p and p+16 halves
// (ISA 16-bit A layout: lane holds K = {kb..kb+7, kb+16..kb+23}).
__device__ __forceinline__ v16h load_a_frag(const _Float16* p) {
  v8h lo = *(const v8h*)p;
  v8h hi = *(const v8h*)(p + 16);
  return __builtin_shufflevector(lo, hi, 0, 1, 2, 3, 4, 5, 6, 7, 8, 9, 10, 11,
                                 12, 13, 14, 15);
}

// B fragment from a K-contiguous (transposed, [N][K]) matrix: K = {kb..kb+7, kb+16..kb+23}.
__device__ __forceinline__ v16h load_b_frag(const _Float16* p) {
  v8h lo = *(const v8h*)p;
  v8h hi = *(const v8h*)(p + 16);
  return __builtin_shufflevector(lo, hi, 0, 1, 2, 3, 4, 5, 6, 7, 8, 9, 10, 11,
                                 12, 13, 14, 15);
}

// PositionalEncoding: even f -> sin(pos*div), odd f -> cos(pos*div)
__device__ __forceinline__ float pe_val(int pos, int f) {
  const float c = -9.210340371976184f / 128.0f;  // -ln(10000)/F_IN
  float ang = (float)pos * __expf((float)(f & ~1) * c);
  return (f & 1) ? __cosf(ang) : __sinf(ang);
}

// ---------------------------------------------------------------------------
// Kernel 0: one-shot f32 -> f16 weight conversion, stored K-contiguous
// (transposed) so WMMA fragments become aligned 16B vector loads.
// ---------------------------------------------------------------------------
__global__ __launch_bounds__(256) void convert_kernel(
    const float* __restrict__ Wq, const float* __restrict__ Wk,
    const float* __restrict__ Wv, const float* __restrict__ Wcls,
    const float* __restrict__ Wft, _Float16* __restrict__ ws) {
  int idx = blockIdx.x * 256 + threadIdx.x;
  if (idx < WQKV_SZ) {  // [mat][col][k] = W[k*512 + col]
    int mat = idx / (512 * 128), rem = idx - mat * (512 * 128);
    int col = rem >> 7, k = rem & 127;
    const float* W = (mat == 0) ? Wq : (mat == 1) ? Wk : Wv;
    ws[WQKV_OFF + idx] = (_Float16)W[k * 512 + col];
  }
  if (idx < WCLS_SZ) {  // [j][k] = Wcls[k*22 + j], zero padded to 32x96
    int m = idx / 96, k = idx - (idx / 96) * 96;
    ws[WCLS_OFF + idx] =
        (_Float16)((m < J_NUM && k < S_LEN) ? Wcls[k * J_NUM + m] : 0.f);
  }
  if (idx < WFT_SZ) {  // [col][k] = Wft[k*512 + col]
    int col = idx >> 9, k = idx & 511;
    ws[WFT_OFF + idx] = (_Float16)Wft[k * 512 + col];
  }
}

// ---------------------------------------------------------------------------
// Kernel 1: one block per (h, w, b). Fused PE + QKV + masked attention + cls.
// ---------------------------------------------------------------------------
__global__ __launch_bounds__(128) void st_attn_kernel(
    const float* __restrict__ x, const float* __restrict__ bq,
    const float* __restrict__ bk, const float* __restrict__ bv,
    const float* __restrict__ bcls, const _Float16* __restrict__ ws,
    _Float16* __restrict__ ybuf) {
  // LDS reuse (57KB):
  //   sA  : xw f16 [80][128] -> scores f32 [80][80] -> attn f16 (stride 160h)
  //   sQK : q f16 [80][64] + k f16 [80][64] -> aoT f16 [64][96]
  //   sV  : vT f16 [64 col][96 row]
  __shared__ float    sA[SP * 80];
  __shared__ _Float16 sQK[SP * 64 * 2];
  __shared__ _Float16 sV[64 * SK];

  const int h = blockIdx.x, w = blockIdx.y, b = blockIdx.z;
  const int tid = threadIdx.x;
  const int wave = tid >> 5, lane = tid & 31;
  const int l15 = lane & 15, lhi = lane >> 4;

  const _Float16* WqkvT = ws + WQKV_OFF;
  const _Float16* WclsT = ws + WCLS_OFF;

  // ---- stage xw = x + Spe + Tpe (rows 66..79 zero) ----
  _Float16* xw = (_Float16*)sA;  // [80][128]
  for (int idx = tid; idx < SP * F_IN; idx += 128) {
    int s = idx >> 7, f = idx & 127;
    float v = 0.f;
    if (s < S_LEN) {
      int fi = s / J_NUM, j = s - fi * J_NUM, t = w + fi;
      v = x[(((size_t)b * T_LEN + t) * J_NUM + j) * F_IN + f] + pe_val(j, f) +
          pe_val(t, f);
    }
    xw[idx] = (_Float16)v;
  }
  // zero vT pad rows 80..95 (NaN-safe K padding for attn@V)
  for (int idx = tid; idx < 64 * 16; idx += 128)
    sV[(idx >> 4) * SK + SP + (idx & 15)] = (_Float16)0.f;
  __syncthreads();

  // ---- Q/K/V = xw @ W[:, h*64:+64] (+bias; ReLU on V); 60 tiles / 4 waves
  for (int tile = wave; tile < 60; tile += 4) {
    int mat = tile / 20, rem = tile - mat * 20;
    int mt = rem >> 2, nt = rem & 3;
    const float* bsel = (mat == 0) ? bq : (mat == 1) ? bk : bv;
    int col = nt * 16 + l15;
    const _Float16* wb = WqkvT + ((size_t)mat * 512 + h * 64 + col) * 128;
    v8f acc = {};
#pragma unroll
    for (int ks = 0; ks < 4; ++ks) {  // K = 128
      v16h a  = load_a_frag(xw + (mt * 16 + l15) * F_IN + ks * 32 + lhi * 8);
      v16h bb = load_b_frag(wb + ks * 32 + lhi * 8);
      acc = wmma16(a, bb, acc);
    }
    float bias = bsel[h * 64 + col];
    if (mat == 2) {  // V stored transposed: vT[col][row]; 8 rows contiguous
      v8h pk;
#pragma unroll
      for (int r = 0; r < 8; ++r) pk[r] = (_Float16)fmaxf(acc[r] + bias, 0.f);
      *(v8h*)(sV + col * SK + mt * 16 + lhi * 8) = pk;
    } else {
      _Float16* dst = (mat == 0) ? sQK : (sQK + SP * 64);
#pragma unroll
      for (int r = 0; r < 8; ++r) {
        int row = mt * 16 + r + lhi * 8;
        dst[row * 64 + col] = (_Float16)(acc[r] + bias);
      }
    }
  }
  __syncthreads();

  // ---- scores = Q @ K^T * 0.125 -> sA f32 [80][80]
  const _Float16* qm = sQK;
  const _Float16* km = sQK + SP * 64;  // B[k][n] = K[n][k]: K-contiguous
  for (int tile = wave; tile < 25; tile += 4) {
    int mt = tile / 5, nt = tile - mt * 5;
    int col = nt * 16 + l15;
    v8f acc = {};
#pragma unroll
    for (int ks = 0; ks < 2; ++ks) {  // K = 64
      v16h a  = load_a_frag(qm + (mt * 16 + l15) * 64 + ks * 32 + lhi * 8);
      v16h bb = load_b_frag(km + col * 64 + ks * 32 + lhi * 8);
      acc = wmma16(a, bb, acc);
    }
#pragma unroll
    for (int r = 0; r < 8; ++r)
      sA[(mt * 16 + r + lhi * 8) * 80 + col] = acc[r] * 0.125f;
  }
  __syncthreads();

  // ---- mask * threshold * softmax, one thread per row; attn f16 written in
  //      place over f32 scores (write idx c/2 <= read idx c already consumed).
  if (tid < S_LEN) {
    float* sc = sA + tid * 80;
    _Float16* at = (_Float16*)sc;  // row stride 160 halves
    const bool rlo = (tid < J_NUM), rhi = (tid >= 2 * J_NUM);
    float m = -3.4e38f;
    for (int c = 0; c < S_LEN; ++c) {
      float v = sc[c];
      if ((rlo && c >= 2 * J_NUM) || (rhi && c < J_NUM)) v = 0.f;
      m = fmaxf(m, v);
    }
    const float thr = m / 9.0f;
    float sum = 0.f;
    for (int c = 0; c < S_LEN; ++c) {
      float v = sc[c];
      if ((rlo && c >= 2 * J_NUM) || (rhi && c < J_NUM)) v = 0.f;
      if (fabsf(v) <= thr) v = NEGV;
      sum += __expf(v - m);
    }
    const float inv = 1.f / sum;
    for (int c = 0; c < S_LEN; ++c) {
      float v = sc[c];
      if ((rlo && c >= 2 * J_NUM) || (rhi && c < J_NUM)) v = 0.f;
      if (fabsf(v) <= thr) v = NEGV;
      at[c] = (_Float16)(__expf(v - m) * inv);
    }
    for (int c = S_LEN; c < SK; ++c) at[c] = (_Float16)0.f;  // K padding
  }
  __syncthreads();

  // ---- aoT = (attn @ V)^T -> sQK reused as f16 [64 col][96 row]
  _Float16* aoT = sQK;
  {
    const _Float16* at = (const _Float16*)sA;  // stride 160 halves
    for (int tile = wave; tile < 20; tile += 4) {
      int mt = tile >> 2, nt = tile & 3;
      int col = nt * 16 + l15;
      v8f acc = {};
#pragma unroll
      for (int ks = 0; ks < 3; ++ks) {  // K = 96
        v16h a  = load_a_frag(at + (mt * 16 + l15) * 160 + ks * 32 + lhi * 8);
        v16h bb = load_b_frag(sV + col * SK + ks * 32 + lhi * 8);
        acc = wmma16(a, bb, acc);
      }
      v8h pk;
#pragma unroll
      for (int r = 0; r < 8; ++r) pk[r] = (_Float16)acc[r];
      *(v8h*)(aoT + col * SK + mt * 16 + lhi * 8) = pk;
    }
  }
  __syncthreads();
  // zero aoT rows 66..95 (pad-row garbage + K padding; NaN-safe)
  if (tid < 64) {
    for (int r = S_LEN; r < SK; ++r) aoT[tid * SK + r] = (_Float16)0.f;
  }
  __syncthreads();

  // ---- cls: D[22x64] = WclsT(22x66, padded) @ ao + bcls -> ybuf (f16)
  for (int tile = wave; tile < 8; tile += 4) {
    int mt = tile >> 2, nt = tile & 3;
    int col = nt * 16 + l15;
    v8f acc = {};
#pragma unroll
    for (int ks = 0; ks < 3; ++ks) {  // K = 96 (zero padded past 66)
      v16h a  = load_a_frag(WclsT + (mt * 16 + l15) * SK + ks * 32 + lhi * 8);
      v16h bb = load_b_frag(aoT + col * SK + ks * 32 + lhi * 8);
      acc = wmma16(a, bb, acc);
    }
#pragma unroll
    for (int r = 0; r < 8; ++r) {
      int j = mt * 16 + r + lhi * 8;
      if (j < J_NUM) {
        float vv = acc[r] + bcls[j];
        size_t rowg = (size_t)b * (W_NUM * J_NUM) + (size_t)w * J_NUM + j;
        ybuf[rowg * 512 + h * 64 + col] = (_Float16)vv;
      }
    }
  }
}

// ---------------------------------------------------------------------------
// TDM: stage a 2D f16 tile (tile_d0 elements contiguous x tile_d1 rows,
// row stride td0_stride elements) from global into LDS. D# per ISA ch.8.
// ---------------------------------------------------------------------------
#if __has_builtin(__builtin_amdgcn_tensor_load_to_lds)
#define HAVE_TDM 1
__device__ __forceinline__ void tdm_load_2d_f16(unsigned lds_off,
                                                unsigned long long gaddr,
                                                unsigned tile_d0,
                                                unsigned tile_d1,
                                                unsigned long long td0_stride) {
  v4u g0;
  g0[0] = 1u;                                   // count=1, user-mode descriptor
  g0[1] = lds_off;                              // lds_addr (bytes)
  g0[2] = (unsigned)(gaddr & 0xffffffffu);      // global_addr[31:0]
  g0[3] = (unsigned)((gaddr >> 32) & 0x01ffffffu) | 0x80000000u;  // type=2
  v8i g1;
  g1[0] = (int)(1u << 16);                      // data_size=1 (2 bytes)
  g1[1] = (int)((tile_d0 & 0xffffu) << 16);     // tensor_dim0 lo16 @ bit48
  g1[2] = (int)(((tile_d0 >> 16) & 0xffffu) | ((tile_d1 & 0xffffu) << 16));
  g1[3] = (int)(((tile_d1 >> 16) & 0xffffu) | ((tile_d0 & 0xffffu) << 16));
  g1[4] = (int)(tile_d1 & 0xffffu);             // tile_dim1; tile_dim2=0
  g1[5] = (int)(td0_stride & 0xffffffffu);      // tensor_dim0_stride lo32
  g1[6] = (int)((td0_stride >> 32) & 0xffffu);  // stride hi16; dim1_stride=0
  g1[7] = 0;
  v4i g2; g2[0] = 1; g2[1] = 1; g2[2] = 0; g2[3] = 0;  // tensor_dim2/3 = 1
  v4i g3; g3[0] = 0; g3[1] = (int)(1u << 16); g3[2] = 0; g3[3] = 0;  // dim4=1
#if defined(__clang_major__) && __clang_major__ >= 23
  v8i g4 = {};
  __builtin_amdgcn_tensor_load_to_lds(g0, g1, g2, g3, g4, 0);
#else
  __builtin_amdgcn_tensor_load_to_lds(g0, g1, g2, g3, 0);
#endif
}
#else
#define HAVE_TDM 0
#endif

// ---------------------------------------------------------------------------
// Kernel 2: y(16896x512) @ Wft + bft, ReLU, LayerNorm(ddof=1, eps on std).
// 16 rows x full 512 cols per 256-thread block. Double-buffered TDM staging
// of WftT K-slices overlapped with WMMA; LayerNorm done from accumulator
// registers (shfl butterfly + 1KB LDS cross-wave reduction) -- no y LDS
// round-trip at all.
// ---------------------------------------------------------------------------
__global__ __launch_bounds__(256) void ft_ln_kernel(
    const _Float16* __restrict__ ws, const float* __restrict__ bft,
    const float* __restrict__ a2, const float* __restrict__ b2,
    float* __restrict__ outp) {
  __shared__ _Float16 sBT[2][512 * 32];  // 2 x 32KB double buffer [col][k]
  // after the K loop, buffer 0 is reused as f32 reduction scratch:
  //   [0..127] row-sum partials [wave][row], [128..255] sumsq partials,
  //   [256..271] mean, [272..287] 1/(std+eps)
  float* scr = (float*)&sBT[0][0];

  const _Float16* ybuf = ws + YB_OFF;
  const _Float16* WftT = ws + WFT_OFF;

  const int tid = threadIdx.x, wave = tid >> 5, lane = tid & 31;
  const int l15 = lane & 15, lhi = lane >> 4;
  const int rowBase = blockIdx.x * 16;
  const int arow = rowBase + l15;

  v8f acc[4] = {};
#if HAVE_TDM
  if (wave == 0) {  // prologue: start slice 0
    tdm_load_2d_f16((unsigned)(size_t)(void*)&sBT[0][0],
                    (unsigned long long)(size_t)(const void*)WftT, 32u, 512u,
                    512ull);
  }
#endif
  for (int ks = 0; ks < 16; ++ks) {  // K = 512
    __syncthreads();  // prior reads of the buffer being (over)written are done
#if HAVE_TDM
    if (wave == 0) {
      if (ks + 1 < 16) {  // overlap: start next slice, then wait current only
        tdm_load_2d_f16((unsigned)(size_t)(void*)&sBT[(ks + 1) & 1][0],
                        (unsigned long long)(size_t)(const void*)WftT +
                            (unsigned long long)(ks + 1) * 64ull,
                        32u, 512u, 512ull);
        __builtin_amdgcn_s_wait_tensorcnt(1);  // in-order: slice ks landed
      } else {
        __builtin_amdgcn_s_wait_tensorcnt(0);
      }
    }
#else
    for (int i = tid; i < 512 * 32; i += 256)
      sBT[ks & 1][i] = WftT[(size_t)(i >> 5) * 512 + ks * 32 + (i & 31)];
#endif
    __syncthreads();
    const _Float16* bufc = &sBT[ks & 1][0];
    // prefetch next A segment while computing this one
    __builtin_prefetch(ybuf + (size_t)arow * 512 + (ks + 1) * 32, 0, 3);
    v16h a = load_a_frag(ybuf + (size_t)arow * 512 + ks * 32 + lhi * 8);
#pragma unroll
    for (int t = 0; t < 4; ++t) {
      int col = (wave + t * 8) * 16 + l15;
      v16h bb = load_b_frag(bufc + col * 32 + lhi * 8);
      acc[t] = wmma16(a, bb, acc[t]);
    }
  }
  __syncthreads();  // all LDS reads done; buffer 0 becomes reduction scratch

  // ---- epilogue in registers: bias + ReLU, per-row partial sums ----
  float vals[4][8];
  float ps[8], ps2[8];
#pragma unroll
  for (int r = 0; r < 8; ++r) { ps[r] = 0.f; ps2[r] = 0.f; }
#pragma unroll
  for (int t = 0; t < 4; ++t) {
    int col = (wave + t * 8) * 16 + l15;
    float bias = bft[col];
#pragma unroll
    for (int r = 0; r < 8; ++r) {
      float v = fmaxf(acc[t][r] + bias, 0.f);
      vals[t][r] = v;
      ps[r] += v;
      ps2[r] += v * v;
    }
  }
  // 16-lane butterfly: lanes 0-15 (resp. 16-31) of a wave hold the same 8
  // rows across 16 columns (f32 C/D layout), so width-16 xor-reduce gives
  // this wave's row sums over its 64 columns.
#pragma unroll
  for (int m = 8; m >= 1; m >>= 1) {
#pragma unroll
    for (int r = 0; r < 8; ++r) {
      ps[r]  += __shfl_xor(ps[r], m, 16);
      ps2[r] += __shfl_xor(ps2[r], m, 16);
    }
  }
  if (l15 == 0) {  // lanes 0 and 16 publish partials for rows lhi*8..+7
#pragma unroll
    for (int r = 0; r < 8; ++r) {
      scr[wave * 16 + lhi * 8 + r]       = ps[r];
      scr[128 + wave * 16 + lhi * 8 + r] = ps2[r];
    }
  }
  __syncthreads();
  if (tid < 16) {  // final cross-wave reduce: one thread per row
    float S = 0.f, S2 = 0.f;
#pragma unroll
    for (int wv = 0; wv < 8; ++wv) {
      S  += scr[wv * 16 + tid];
      S2 += scr[128 + wv * 16 + tid];
    }
    float mean = S * (1.0f / 512.0f);
    float var  = fmaxf((S2 - S * S * (1.0f / 512.0f)) * (1.0f / 511.0f), 0.f);
    scr[256 + tid] = mean;
    scr[272 + tid] = 1.0f / (sqrtf(var) + 1e-6f);  // eps added to std (ref)
  }
  __syncthreads();
  // ---- normalize from registers and store straight to global ----
#pragma unroll
  for (int r = 0; r < 8; ++r) {
    int row = r + lhi * 8;
    float mean = scr[256 + row];
    float inv  = scr[272 + row];
    size_t gRow = (size_t)rowBase + row;
#pragma unroll
    for (int t = 0; t < 4; ++t) {
      int col = (wave + t * 8) * 16 + l15;
      outp[gRow * 512 + col] = a2[col] * (vals[t][r] - mean) * inv + b2[col];
    }
  }
}

extern "C" void kernel_launch(void* const* d_in, const int* in_sizes, int n_in,
                              void* d_out, int out_size, void* d_ws, size_t ws_size,
                              hipStream_t stream) {
  (void)in_sizes; (void)n_in; (void)out_size; (void)ws_size;
  const float* x    = (const float*)d_in[0];
  const float* Wq   = (const float*)d_in[1];
  const float* bq   = (const float*)d_in[2];
  const float* Wk   = (const float*)d_in[3];
  const float* bk   = (const float*)d_in[4];
  const float* Wv   = (const float*)d_in[5];
  const float* bv   = (const float*)d_in[6];
  const float* Wcls = (const float*)d_in[7];
  const float* bcls = (const float*)d_in[8];
  const float* Wft  = (const float*)d_in[9];
  const float* bft  = (const float*)d_in[10];
  const float* a2   = (const float*)d_in[11];
  const float* b2   = (const float*)d_in[12];
  _Float16* ws = (_Float16*)d_ws;  // ~17.4 MB used

  convert_kernel<<<(WFT_SZ + 255) / 256, 256, 0, stream>>>(Wq, Wk, Wv, Wcls,
                                                           Wft, ws);
  dim3 g1(H_NUM, W_NUM, B_SZ);  // 6144 blocks x 4 waves
  st_attn_kernel<<<g1, 128, 0, stream>>>(x, bq, bk, bv, bcls, ws, ws + YB_OFF);
  const int rows = B_SZ * W_NUM * J_NUM;  // 16896
  ft_ln_kernel<<<rows / 16, 256, 0, stream>>>(ws, bft, a2, b2, (float*)d_out);
}
